// MambaBlock_10763188043928
// MI455X (gfx1250) — compile-verified, matrix operand order checked by reading, not run on hardware
//
#include <hip/hip_runtime.h>
#include <hip/hip_bf16.h>

typedef __bf16 bf16;
typedef __attribute__((ext_vector_type(16))) __bf16 v16bf;
typedef __attribute__((ext_vector_type(8)))  float  v8f;

union FragU { v16bf v; uint4 q[2]; };

#define WMMA_BF16(a, b, c) \
  __builtin_amdgcn_wmma_f32_16x16x32_bf16(false, (a), false, (b), (short)0, (c), false, false)

// ---------------------------------------------------------------------------
// Fragment loaders matching CDNA5 16-bit WMMA VGPR layout (ISA 7.12.2):
// lane = 16*half + r ; element e<8  -> K = k0 + 8*half + e
//                      element e>=8 -> K = k0 + 16 + 8*half + (e-8)
// Works for A (row = M) and for B when the weight is stored transposed [N][K].
// ---------------------------------------------------------------------------
static __device__ __forceinline__ v16bf load_frag(const bf16* __restrict__ base,
                                                  int ld, int row0, int k0) {
  const int lane = threadIdx.x & 31;
  const int half = lane >> 4;
  const int r    = lane & 15;
  const bf16* p = base + (size_t)(row0 + r) * ld + (k0 + 8 * half);
  FragU f;
  f.q[0] = *reinterpret_cast<const uint4*>(p);
  f.q[1] = *reinterpret_cast<const uint4*>(p + 16);
  return f.v;
}

// Rows with r < rmin are zeroed (causal-conv left edge, avoids OOB deref).
static __device__ __forceinline__ v16bf load_frag_rmin(const bf16* __restrict__ base,
                                                       int ld, int row0, int k0, int rmin) {
  const int lane = threadIdx.x & 31;
  const int half = lane >> 4;
  const int r    = lane & 15;
  FragU f;
  if (r >= rmin) {
    const bf16* p = base + (size_t)(row0 + r) * ld + (k0 + 8 * half);
    f.q[0] = *reinterpret_cast<const uint4*>(p);
    f.q[1] = *reinterpret_cast<const uint4*>(p + 16);
  } else {
    f.q[0] = make_uint4(0u, 0u, 0u, 0u);
    f.q[1] = make_uint4(0u, 0u, 0u, 0u);
  }
  return f.v;
}

// K dimension is only 16 wide: upper half (K>=16) zero-padded.
static __device__ __forceinline__ v16bf load_frag_k16(const bf16* __restrict__ base,
                                                      int ld, int row0) {
  const int lane = threadIdx.x & 31;
  const int half = lane >> 4;
  const int r    = lane & 15;
  const bf16* p = base + (size_t)(row0 + r) * ld + 8 * half;
  FragU f;
  f.q[0] = *reinterpret_cast<const uint4*>(p);
  f.q[1] = make_uint4(0u, 0u, 0u, 0u);
  return f.v;
}

static __device__ __forceinline__ float silu_f(float s) {
  return s / (1.f + __expf(-s));
}
static __device__ __forceinline__ float softplus_f(float s) {
  return (s > 20.f) ? s : log1pf(__expf(s));
}

// ---------------------------------------------------------------------------
// K0: bf16 converts, weight transposes, A = -exp(A_log)
// ---------------------------------------------------------------------------
__global__ void k_prep(const float* __restrict__ x, const float* __restrict__ W_in,
                       const float* __restrict__ W_res, const float* __restrict__ conv_w,
                       const float* __restrict__ W_xdt, const float* __restrict__ W_dt,
                       const float* __restrict__ W_B, const float* __restrict__ W_C,
                       const float* __restrict__ A_log, const float* __restrict__ W_out,
                       bf16* x_bf, bf16* W_inT, bf16* W_resT, bf16* convT,
                       bf16* W_xdtT, bf16* W_dtT, bf16* W_BT, bf16* W_CT,
                       bf16* W_outT, float* A_neg) {
  const int stride = gridDim.x * blockDim.x;
  const int gid = blockIdx.x * blockDim.x + threadIdx.x;
  for (int i = gid; i < 1024 * 128; i += stride) x_bf[i] = (bf16)x[i];
  for (int i = gid; i < 256 * 128; i += stride) {           // [n<256][k<128]
    int n = i >> 7, k = i & 127;
    W_inT[i]  = (bf16)W_in[k * 256 + n];
    W_resT[i] = (bf16)W_res[k * 256 + n];
  }
  for (int i = gid; i < 4 * 256 * 256; i += stride) {       // [t][o][i]
    int t = i >> 16, rem = i & 65535, o = rem >> 8, ii = rem & 255;
    convT[i] = (bf16)conv_w[o * 1024 + ii * 4 + t];
  }
  for (int i = gid; i < 16 * 256; i += stride) {            // [r<16][k<256]
    int r = i >> 8, k = i & 255;
    W_xdtT[i] = (bf16)W_xdt[k * 16 + r];
  }
  for (int i = gid; i < 256 * 16; i += stride) {            // [n<256][r<16]
    int n = i >> 4, r = i & 15;
    W_dtT[i] = (bf16)W_dt[r * 256 + n];
  }
  for (int i = gid; i < 256 * 256; i += stride) {           // [n<256][k<256]
    int n = i >> 8, k = i & 255;
    W_BT[i] = (bf16)W_B[k * 256 + n];
    W_CT[i] = (bf16)W_C[k * 256 + n];
    A_neg[i] = -__expf(A_log[i]);
  }
  for (int i = gid; i < 128 * 256; i += stride) {           // [n<128][k<256]
    int n = i >> 8, k = i & 255;
    W_outT[i] = (bf16)W_out[k * 128 + n];
  }
}

// ---------------------------------------------------------------------------
// K1: xi = x@W_in + b_in (bf16) ; x_res = silu(x@W_res + b_res) (f32)
// grid (64, 4, 2), block 32. Software-pipelined K loop (4 chunks).
// ---------------------------------------------------------------------------
__global__ void k_proj(const bf16* __restrict__ x_bf, const bf16* __restrict__ W_inT,
                       const bf16* __restrict__ W_resT, const float* __restrict__ b_in,
                       const float* __restrict__ b_res, bf16* __restrict__ xi_bf,
                       float* __restrict__ x_res) {
  const int row0 = blockIdx.x * 16;
  const int col0 = blockIdx.y * 64;
  const int z = blockIdx.z;
  const bf16* WT = z ? W_resT : W_inT;
  v8f acc[4] = {};
  v16bf a = load_frag(x_bf, 128, row0, 0);
  v16bf bq[4];
#pragma unroll
  for (int j = 0; j < 4; ++j) bq[j] = load_frag(WT, 128, col0 + 16 * j, 0);
#pragma unroll
  for (int kc = 0; kc < 3; ++kc) {
    v16bf an = load_frag(x_bf, 128, row0, (kc + 1) * 32);
    v16bf bn[4];
#pragma unroll
    for (int j = 0; j < 4; ++j) bn[j] = load_frag(WT, 128, col0 + 16 * j, (kc + 1) * 32);
#pragma unroll
    for (int j = 0; j < 4; ++j) acc[j] = WMMA_BF16(a, bq[j], acc[j]);
    a = an;
#pragma unroll
    for (int j = 0; j < 4; ++j) bq[j] = bn[j];
  }
#pragma unroll
  for (int j = 0; j < 4; ++j) acc[j] = WMMA_BF16(a, bq[j], acc[j]);

  const int lane = threadIdx.x & 31, half = lane >> 4, c = lane & 15;
#pragma unroll
  for (int j = 0; j < 4; ++j)
#pragma unroll
    for (int v = 0; v < 8; ++v) {
      const int row = row0 + v + 8 * half;
      const int col = col0 + 16 * j + c;
      if (z == 0) {
        xi_bf[row * 256 + col] = (bf16)(acc[j][v] + b_in[col]);
      } else {
        x_res[row * 256 + col] = silu_f(acc[j][v] + b_res[col]);
      }
    }
}

// ---------------------------------------------------------------------------
// K2: causal full conv1d (K=4) as 4 shifted GEMMs; u = silu(. + conv_b)
// grid (64, 4), block 32. Flattened 32-chunk pipelined loop.
// ---------------------------------------------------------------------------
__global__ void k_conv(const bf16* __restrict__ xi_bf, const bf16* __restrict__ convT,
                       const float* __restrict__ conv_b, float* __restrict__ u_f,
                       bf16* __restrict__ u_bf) {
  const int row0 = blockIdx.x * 16;
  const int col0 = blockIdx.y * 64;
  const int lbase = row0 & 511;
  v8f acc[4] = {};

  auto loadA = [&](int m) -> v16bf {
    const int t = m >> 3, kc = m & 7;
    const int shift = t - 3;
    return load_frag_rmin(xi_bf, 256, row0 + shift, kc * 32, -shift - lbase);
  };
  auto loadB = [&](int m, int j) -> v16bf {
    const int t = m >> 3, kc = m & 7;
    return load_frag(convT + t * 65536, 256, col0 + 16 * j, kc * 32);
  };

  v16bf a = loadA(0);
  v16bf bq[4];
#pragma unroll
  for (int j = 0; j < 4; ++j) bq[j] = loadB(0, j);
  for (int m = 0; m < 31; ++m) {
    v16bf an = loadA(m + 1);
    v16bf bn[4];
#pragma unroll
    for (int j = 0; j < 4; ++j) bn[j] = loadB(m + 1, j);
#pragma unroll
    for (int j = 0; j < 4; ++j) acc[j] = WMMA_BF16(a, bq[j], acc[j]);
    a = an;
#pragma unroll
    for (int j = 0; j < 4; ++j) bq[j] = bn[j];
  }
#pragma unroll
  for (int j = 0; j < 4; ++j) acc[j] = WMMA_BF16(a, bq[j], acc[j]);

  const int lane = threadIdx.x & 31, half = lane >> 4, c = lane & 15;
#pragma unroll
  for (int j = 0; j < 4; ++j)
#pragma unroll
    for (int v = 0; v < 8; ++v) {
      const int row = row0 + v + 8 * half;
      const int col = col0 + 16 * j + c;
      const float uv = silu_f(acc[j][v] + conv_b[col]);
      u_f[row * 256 + col] = uv;
      u_bf[row * 256 + col] = (bf16)uv;
    }
}

// ---------------------------------------------------------------------------
// K3: tmp = u @ W_xdt  -> [BL,16] bf16.  grid (64), block 32. Pipelined.
// ---------------------------------------------------------------------------
__global__ void k_xdt(const bf16* __restrict__ u_bf, const bf16* __restrict__ W_xdtT,
                      bf16* __restrict__ tmp_bf) {
  const int row0 = blockIdx.x * 16;
  v8f acc = {};
  v16bf a = load_frag(u_bf, 256, row0, 0);
  v16bf bq = load_frag(W_xdtT, 256, 0, 0);
#pragma unroll
  for (int kc = 0; kc < 7; ++kc) {
    v16bf an = load_frag(u_bf, 256, row0, (kc + 1) * 32);
    v16bf bn = load_frag(W_xdtT, 256, 0, (kc + 1) * 32);
    acc = WMMA_BF16(a, bq, acc);
    a = an;
    bq = bn;
  }
  acc = WMMA_BF16(a, bq, acc);
  const int lane = threadIdx.x & 31, half = lane >> 4, c = lane & 15;
#pragma unroll
  for (int v = 0; v < 8; ++v)
    tmp_bf[(row0 + v + 8 * half) * 16 + c] = (bf16)acc[v];
}

// ---------------------------------------------------------------------------
// K4: delta = softplus(tmp @ W_dt + b_dt)  (K=16 zero-padded WMMA)
// grid (64, 4), block 32.
// ---------------------------------------------------------------------------
__global__ void k_dt(const bf16* __restrict__ tmp_bf, const bf16* __restrict__ W_dtT,
                     const float* __restrict__ b_dt, float* __restrict__ delta) {
  const int row0 = blockIdx.x * 16;
  const int col0 = blockIdx.y * 64;
  v16bf a = load_frag_k16(tmp_bf, 16, row0);
  v8f acc[4] = {};
#pragma unroll
  for (int j = 0; j < 4; ++j) {
    v16bf b = load_frag_k16(W_dtT, 16, col0 + 16 * j);
    acc[j] = WMMA_BF16(a, b, acc[j]);
  }
  const int lane = threadIdx.x & 31, half = lane >> 4, c = lane & 15;
#pragma unroll
  for (int j = 0; j < 4; ++j)
#pragma unroll
    for (int v = 0; v < 8; ++v) {
      const int row = row0 + v + 8 * half;
      const int col = col0 + 16 * j + c;
      delta[row * 256 + col] = softplus_f(acc[j][v] + b_dt[col]);
    }
}

// ---------------------------------------------------------------------------
// K5: Bm = u@W_B, Cm = u@W_C (f32 out).  grid (64, 4, 2), block 32. Pipelined.
// ---------------------------------------------------------------------------
__global__ void k_bc(const bf16* __restrict__ u_bf, const bf16* __restrict__ W_BT,
                     const bf16* __restrict__ W_CT, float* __restrict__ Bm,
                     float* __restrict__ Cm) {
  const int row0 = blockIdx.x * 16;
  const int col0 = blockIdx.y * 64;
  const bf16* WT = blockIdx.z ? W_CT : W_BT;
  float* out = blockIdx.z ? Cm : Bm;
  v8f acc[4] = {};
  v16bf a = load_frag(u_bf, 256, row0, 0);
  v16bf bq[4];
#pragma unroll
  for (int j = 0; j < 4; ++j) bq[j] = load_frag(WT, 256, col0 + 16 * j, 0);
  for (int kc = 0; kc < 7; ++kc) {
    v16bf an = load_frag(u_bf, 256, row0, (kc + 1) * 32);
    v16bf bn[4];
#pragma unroll
    for (int j = 0; j < 4; ++j) bn[j] = load_frag(WT, 256, col0 + 16 * j, (kc + 1) * 32);
#pragma unroll
    for (int j = 0; j < 4; ++j) acc[j] = WMMA_BF16(a, bq[j], acc[j]);
    a = an;
#pragma unroll
    for (int j = 0; j < 4; ++j) bq[j] = bn[j];
  }
#pragma unroll
  for (int j = 0; j < 4; ++j) acc[j] = WMMA_BF16(a, bq[j], acc[j]);

  const int lane = threadIdx.x & 31, half = lane >> 4, c = lane & 15;
#pragma unroll
  for (int j = 0; j < 4; ++j)
#pragma unroll
    for (int v = 0; v < 8; ++v)
      out[(row0 + v + 8 * half) * 256 + col0 + 16 * j + c] = acc[j][v];
}

// ---------------------------------------------------------------------------
// K6: selective scan, ping-pong double-buffered CDNA5 async-DMA staging.
// block 256 (8 waves): wave owns a d-row, lane owns 8 n's. Iteration l issues
// GLOBAL_LOAD_ASYNC_TO_LDS for step l+1 into the alternate LDS buffer, then
// s_wait_asynccnt 0x2 retires only the *previous* pair (in-order completion),
// so the DMA overlaps the exp/FMA chain and the wave32 shfl reduction.
// delta/u scalars get the same one-step register double-buffer.
// grid (64): blocks 0..31 -> batch 0, 32..63 -> batch 1.
// ---------------------------------------------------------------------------
__global__ __launch_bounds__(256) void k_scan(const float* __restrict__ delta,
                                              const float* __restrict__ u_f,
                                              const float* __restrict__ Bm,
                                              const float* __restrict__ Cm,
                                              const float* __restrict__ A_neg,
                                              const float* __restrict__ Dv,
                                              const float* __restrict__ x_res,
                                              bf16* __restrict__ y_bf) {
  __shared__ float sB[2][256];
  __shared__ float sC[2][256];
  const int b = blockIdx.x >> 5;
  const int d0 = (blockIdx.x & 31) * 8;
  const int wave = threadIdx.x >> 5;
  const int lane = threadIdx.x & 31;
  const int d = d0 + wave;
  float Areg[8], h[8];
#pragma unroll
  for (int j = 0; j < 8; ++j) {
    Areg[j] = A_neg[d * 256 + lane + 32 * j];
    h[j] = 0.f;
  }
  const float Dd = Dv[d];
  // LDS byte addresses (generic shared pointer low 32 bits == LDS address)
  const unsigned ldsB0 = (unsigned)(size_t)(&sB[0][threadIdx.x]);
  const unsigned ldsB1 = (unsigned)(size_t)(&sB[1][threadIdx.x]);
  const unsigned ldsC0 = (unsigned)(size_t)(&sC[0][threadIdx.x]);
  const unsigned ldsC1 = (unsigned)(size_t)(&sC[1][threadIdx.x]);
  const unsigned voff = (unsigned)threadIdx.x * 4u;
  const size_t base0 = (size_t)b * 512 * 256;

  // Prologue: stage step 0 into buffer 0; preload step-0 scalars.
  asm volatile("global_load_async_to_lds_b32 %0, %1, %2"
               :: "v"(ldsB0), "v"(voff), "s"(Bm + base0) : "memory");
  asm volatile("global_load_async_to_lds_b32 %0, %1, %2"
               :: "v"(ldsC0), "v"(voff), "s"(Cm + base0) : "memory");
  float dt_c = delta[base0 + d];
  float uv_c = u_f[base0 + d];

  for (int l = 0; l < 512; ++l) {
    const int bl = b * 512 + l;
    const int lp1 = (l < 511) ? (l + 1) : 511;     // clamped: keeps wait imm constant
    const size_t rn = ((size_t)b * 512 + lp1) * 256;
    const unsigned nbB = ((l + 1) & 1) ? ldsB1 : ldsB0;
    const unsigned nbC = ((l + 1) & 1) ? ldsC1 : ldsC0;
    // stage next step's B/C rows into the alternate buffer (overlaps compute)
    asm volatile("global_load_async_to_lds_b32 %0, %1, %2"
                 :: "v"(nbB), "v"(voff), "s"(Bm + rn) : "memory");
    asm volatile("global_load_async_to_lds_b32 %0, %1, %2"
                 :: "v"(nbC), "v"(voff), "s"(Cm + rn) : "memory");
    // warm near cache a few steps out
    if (l + 4 < 512) {
      __builtin_prefetch(Bm + (size_t)(bl + 4) * 256 + threadIdx.x, 0, 0);
      __builtin_prefetch(Cm + (size_t)(bl + 4) * 256 + threadIdx.x, 0, 0);
    }
    // next step's scalar operands (register double-buffer, overlaps compute)
    const float dt_n = delta[rn + d];
    const float uv_n = u_f[rn + d];
    // retire the pair issued last iteration; the 2 just-issued may stay in flight
    asm volatile("s_wait_asynccnt 0x2" ::: "memory");
    __syncthreads();
    const float* __restrict__ Bcur = sB[l & 1];
    const float* __restrict__ Ccur = sC[l & 1];
    const float du = dt_c * uv_c;
    float acc = 0.f;
#pragma unroll
    for (int j = 0; j < 8; ++j) {
      const int n = lane + 32 * j;
      const float hn = __expf(dt_c * Areg[j]) * h[j] + du * Bcur[n];
      h[j] = hn;
      acc += hn * Ccur[n];
    }
#pragma unroll
    for (int off = 16; off > 0; off >>= 1) acc += __shfl_xor(acc, off, 32);
    if (lane == 0) {
      const float yt = acc + uv_c * Dd + x_res[bl * 256 + d];
      y_bf[bl * 256 + d] = (bf16)yt;
    }
    __syncthreads();   // protects buffer (l&1) from being overwritten at l+1's issue
    dt_c = dt_n;
    uv_c = uv_n;
  }
}

// ---------------------------------------------------------------------------
// K7: out = y_total @ W_out + b_out -> d_out (f32 [2,512,128])
// grid (64, 2), block 32. Pipelined.
// ---------------------------------------------------------------------------
__global__ void k_out(const bf16* __restrict__ y_bf, const bf16* __restrict__ W_outT,
                      const float* __restrict__ b_out, float* __restrict__ out) {
  const int row0 = blockIdx.x * 16;
  const int col0 = blockIdx.y * 64;
  v8f acc[4] = {};
  v16bf a = load_frag(y_bf, 256, row0, 0);
  v16bf bq[4];
#pragma unroll
  for (int j = 0; j < 4; ++j) bq[j] = load_frag(W_outT, 256, col0 + 16 * j, 0);
  for (int kc = 0; kc < 7; ++kc) {
    v16bf an = load_frag(y_bf, 256, row0, (kc + 1) * 32);
    v16bf bn[4];
#pragma unroll
    for (int j = 0; j < 4; ++j) bn[j] = load_frag(W_outT, 256, col0 + 16 * j, (kc + 1) * 32);
#pragma unroll
    for (int j = 0; j < 4; ++j) acc[j] = WMMA_BF16(a, bq[j], acc[j]);
    a = an;
#pragma unroll
    for (int j = 0; j < 4; ++j) bq[j] = bn[j];
  }
#pragma unroll
  for (int j = 0; j < 4; ++j) acc[j] = WMMA_BF16(a, bq[j], acc[j]);

  const int lane = threadIdx.x & 31, half = lane >> 4, c = lane & 15;
#pragma unroll
  for (int j = 0; j < 4; ++j)
#pragma unroll
    for (int v = 0; v < 8; ++v) {
      const int row = row0 + v + 8 * half;
      const int col = col0 + 16 * j + c;
      out[row * 128 + col] = acc[j][v] + b_out[col];
    }
}

// ---------------------------------------------------------------------------
extern "C" void kernel_launch(void* const* d_in, const int* in_sizes, int n_in,
                              void* d_out, int out_size, void* d_ws, size_t ws_size,
                              hipStream_t stream) {
  (void)in_sizes; (void)n_in; (void)out_size; (void)ws_size;
  const float* x      = (const float*)d_in[0];
  const float* W_in   = (const float*)d_in[1];
  const float* b_in   = (const float*)d_in[2];
  const float* W_res  = (const float*)d_in[3];
  const float* b_res  = (const float*)d_in[4];
  const float* conv_w = (const float*)d_in[5];
  const float* conv_b = (const float*)d_in[6];
  const float* W_xdt  = (const float*)d_in[7];
  const float* W_dt   = (const float*)d_in[8];
  const float* b_dt   = (const float*)d_in[9];
  const float* W_B    = (const float*)d_in[10];
  const float* W_C    = (const float*)d_in[11];
  const float* A_log  = (const float*)d_in[12];
  const float* Dv     = (const float*)d_in[13];
  const float* W_out  = (const float*)d_in[14];
  const float* b_out  = (const float*)d_in[15];
  float* out = (float*)d_out;

  char* ws = (char*)d_ws;
  size_t off = 0;
  auto take = [&](size_t bytes) -> char* {
    char* p = ws + off;
    off = (off + bytes + 255) & ~(size_t)255;
    return p;
  };
  bf16*  x_bf   = (bf16*)take(1024 * 128 * 2);
  bf16*  W_inT  = (bf16*)take(256 * 128 * 2);
  bf16*  W_resT = (bf16*)take(256 * 128 * 2);
  bf16*  convT  = (bf16*)take(4 * 256 * 256 * 2);
  bf16*  W_xdtT = (bf16*)take(16 * 256 * 2);
  bf16*  W_dtT  = (bf16*)take(256 * 16 * 2);
  bf16*  W_BT   = (bf16*)take(256 * 256 * 2);
  bf16*  W_CT   = (bf16*)take(256 * 256 * 2);
  bf16*  W_outT = (bf16*)take(128 * 256 * 2);
  float* A_neg  = (float*)take(256 * 256 * 4);
  bf16*  xi_bf  = (bf16*)take(1024 * 256 * 2);
  float* x_res  = (float*)take(1024 * 256 * 4);
  float* u_f    = (float*)take(1024 * 256 * 4);
  bf16*  u_bf   = (bf16*)take(1024 * 256 * 2);
  bf16*  tmp_bf = (bf16*)take(1024 * 16 * 2);
  float* delta  = (float*)take(1024 * 256 * 4);
  float* Bm     = (float*)take(1024 * 256 * 4);
  float* Cm     = (float*)take(1024 * 256 * 4);
  bf16*  y_bf   = (bf16*)take(1024 * 256 * 2);

  k_prep<<<dim3(512), dim3(256), 0, stream>>>(x, W_in, W_res, conv_w, W_xdt, W_dt,
                                              W_B, W_C, A_log, W_out, x_bf, W_inT,
                                              W_resT, convT, W_xdtT, W_dtT, W_BT,
                                              W_CT, W_outT, A_neg);
  k_proj<<<dim3(64, 4, 2), dim3(32), 0, stream>>>(x_bf, W_inT, W_resT, b_in, b_res,
                                                  xi_bf, x_res);
  k_conv<<<dim3(64, 4), dim3(32), 0, stream>>>(xi_bf, convT, conv_b, u_f, u_bf);
  k_xdt<<<dim3(64), dim3(32), 0, stream>>>(u_bf, W_xdtT, tmp_bf);
  k_dt<<<dim3(64, 4), dim3(32), 0, stream>>>(tmp_bf, W_dtT, b_dt, delta);
  k_bc<<<dim3(64, 4, 2), dim3(32), 0, stream>>>(u_bf, W_BT, W_CT, Bm, Cm);
  k_scan<<<dim3(64), dim3(256), 0, stream>>>(delta, u_f, Bm, Cm, A_neg, Dv, x_res,
                                             y_bf);
  k_out<<<dim3(64, 2), dim3(32), 0, stream>>>(y_bf, W_outT, b_out, out);
}
